// PerformerAttention_19834158973059
// MI455X (gfx1250) — matrix operand order repaired and checked by reading, not compiled, not run on hardware
//
#include <hip/hip_runtime.h>
#include <hip/hip_bf16.h>

// Performer attention for MI455X (gfx1250), bf16 WMMA path.
// B=4 H=16 N=4096 D=64 R=256.

#define B_   4
#define H_   16
#define NN   4096
#define DD   64
#define RR   256
#define D2   80     // 64 V columns + ones column (64) + pad; only cols 0..64 meaningful
#define NT   128    // rows per n-tile
#define BH   (B_*H_)

typedef __attribute__((ext_vector_type(16))) __bf16 v16bf;
typedef __attribute__((ext_vector_type(8)))  __bf16 v8bf;
typedef __attribute__((ext_vector_type(8)))  float  v8f;

// LDS row strides (elements). All are multiples of 8 (16B alignment for
// v8bf/b128 accesses) and chosen so row-to-row bank offset != 0 mod 64 dwords.
#define PT_S   72    // P^T  [RR][72]   (144B rows)
#define XT_S   72    // Q/K  [NT][72]
#define KPT_S  136   // Kp^T [RR][136]  (272B rows)
#define VT_S   136   // V^T  [DD][136]
#define QP_S   264   // Qp   [NT][264]  (528B rows)
#define KVT_S  264   // KV^T [65][264]

__device__ __forceinline__ v8f wmma_bf16(v16bf a, v16bf b, v8f c) {
  return __builtin_amdgcn_wmma_f32_16x16x32_bf16(false, a, false, b, (short)0, c,
                                                 false, false);
}

// A-fragment (16x32 bf16): lane in [0,16): K = {0..15}; lane in [16,32):
// K = {16..31}. p points at row base offset kb*32; klo = (lane<16)?0:8.
__device__ __forceinline__ v16bf make_a_frag(const __bf16* p, int klo) {
  v8bf lo = *(const v8bf*)(p + 2 * klo);
  v8bf hi = *(const v8bf*)(p + 2 * klo + 8);
  return __builtin_shufflevector(lo, hi, 0, 1, 2, 3, 4, 5, 6, 7,
                                 8, 9, 10, 11, 12, 13, 14, 15);
}

// scale by 1/sqrt(64), clamp to [-10,10], exp, convert to bf16
__device__ __forceinline__ v8bf feat8(v8f c) {
  v8bf r;
#pragma unroll
  for (int v = 0; v < 8; ++v) {
    float x = c[v] * 0.125f;
    x = fminf(10.0f, fmaxf(-10.0f, x));
    r[v] = (__bf16)__expf(x);
  }
  return r;
}

__global__ void perf_zero_kernel(float* __restrict__ p, int n) {
  int i = blockIdx.x * 256 + threadIdx.x;
  if (i < n) p[i] = 0.0f;
}

// ---------------------------------------------------------------------------
// Kernel 1: KV[bh][r][d2] += sum_n exp(clip(K·P/8))[n,r] * Vx[n,d2]
// where Vx has a virtual ones-column at d2==64 (=> KV[:,64] = ksum).
// grid (BH, 8); each block handles 4 chunks of 128 rows; f32 atomics into ws.
// ---------------------------------------------------------------------------
__global__ __launch_bounds__(256)
void perf_kv_kernel(const float* __restrict__ Kg, const float* __restrict__ Vg,
                    const float* __restrict__ Pg, float* __restrict__ KVws) {
  __shared__ __bf16 PT[RR][PT_S];    // P^T  [r][d]
  __shared__ __bf16 XT[NT][XT_S];    // K    [n][d]
  __shared__ __bf16 KPT[RR][KPT_S];  // Kp^T [r][n]
  __shared__ __bf16 VT[DD][VT_S];    // V^T  [d][n]

  const int tid = threadIdx.x;
  const int lane = tid & 31;
  const int w = tid >> 5;        // 8 waves (wave32)
  const int ml = lane & 15;
  const int hi = lane >> 4;
  const int klo = hi ? 8 : 0;
  const int bh = blockIdx.x;

  // stage P^T (coalesced global reads; row r = tid)
  for (int d = 0; d < DD; ++d) PT[tid][d] = (__bf16)Pg[d * RR + tid];

  // constant B-fragment for the d-tile holding the ones column:
  // lane column (ml==0) is all ones, other columns are zero.
  v16bf bones;
#pragma unroll
  for (int v = 0; v < 16; ++v) bones[v] = (ml == 0) ? (__bf16)1.0f : (__bf16)0.0f;

  v8f acc[2][5];
  v8f vzero = 0.0f;
#pragma unroll
  for (int i = 0; i < 2; ++i)
#pragma unroll
    for (int dt = 0; dt < 5; ++dt) acc[i][dt] = vzero;

  const float* Kb = Kg + (size_t)bh * NN * DD;
  const float* Vb = Vg + (size_t)bh * NN * DD;

  for (int j = 0; j < 4; ++j) {
    const int n0 = (blockIdx.y * 4 + j) * NT;
    __syncthreads();  // previous iteration done reading LDS
    for (int i = tid; i < NT * DD; i += 256) {
      int n = i / DD, d = i % DD;
      float kv = Kb[(size_t)(n0 + n) * DD + d];
      float vv = Vb[(size_t)(n0 + n) * DD + d];
      XT[n][d] = (__bf16)kv;
      VT[d][n] = (__bf16)vv;
    }
    __syncthreads();

    // feature matmul: C[n,r] tile strip for rows w*16..w*16+15 (M=n, N=r)
    {
      const __bf16* arow = &XT[w * 16 + ml][0];
      v16bf a0 = make_a_frag(arow + 0, klo);
      v16bf a1 = make_a_frag(arow + 32, klo);
      for (int rt = 0; rt < 16; ++rt) {
        const __bf16* brow = &PT[rt * 16 + ml][0];
        v16bf b0 = *(const v16bf*)(brow + hi * 16);
        v16bf b1 = *(const v16bf*)(brow + 32 + hi * 16);
        v8f c = vzero;
        c = wmma_bf16(a0, b0, c);
        c = wmma_bf16(a1, b1, c);
        v8bf f = feat8(c);
        // lane holds r = rt*16+ml fixed, n = w*16 + 8*hi + v contiguous
        *(v8bf*)&KPT[rt * 16 + ml][w * 16 + 8 * hi] = f;
      }
    }
    __syncthreads();

    // KV += Kp^T @ Vx : M=r (tiles 2w, 2w+1), N=d2 (5 tiles), K = 128
#pragma unroll
    for (int kb = 0; kb < 4; ++kb) {
      v16bf ka0 = make_a_frag(&KPT[(2 * w + 0) * 16 + ml][kb * 32], klo);
      v16bf ka1 = make_a_frag(&KPT[(2 * w + 1) * 16 + ml][kb * 32], klo);
#pragma unroll
      for (int dt = 0; dt < 5; ++dt) {
        v16bf vb = (dt < 4)
                       ? *(const v16bf*)&VT[dt * 16 + ml][kb * 32 + hi * 16]
                       : bones;
        acc[0][dt] = wmma_bf16(ka0, vb, acc[0][dt]);
        acc[1][dt] = wmma_bf16(ka1, vb, acc[1][dt]);
      }
    }
  }

  // accumulate into workspace KV[bh][r][d2]
  float* kvb = KVws + (size_t)bh * RR * D2;
#pragma unroll
  for (int i = 0; i < 2; ++i) {
    int r0 = (2 * w + i) * 16 + 8 * hi;
#pragma unroll
    for (int dt = 0; dt < 5; ++dt) {
      if (dt == 4 && ml != 0) continue;  // only ones-column is meaningful
#pragma unroll
      for (int v = 0; v < 8; ++v) {
        unsafeAtomicAdd(&kvb[(size_t)(r0 + v) * D2 + dt * 16 + ml],
                        acc[i][dt][v]);
      }
    }
  }
}

// ---------------------------------------------------------------------------
// Kernel 2: out[n,d] = (Qp @ KV)[n,d] / max((Qp @ ksum)[n], 1e-6)
// grid (BH, NN/NT).
// ---------------------------------------------------------------------------
__global__ __launch_bounds__(256)
void perf_out_kernel(const float* __restrict__ Qg, const float* __restrict__ Pg,
                     const float* __restrict__ KVws, float* __restrict__ Og) {
  __shared__ __bf16 PT[RR][PT_S];     // P^T  [r][d]
  __shared__ __bf16 XT[NT][XT_S];     // Q    [n][d]
  __shared__ __bf16 QP[NT][QP_S];     // Qp   [n][r]
  __shared__ __bf16 KVT[65][KVT_S];   // KV^T [d2][r], row 64 = ksum
  __shared__ float NRM[NT];

  const int tid = threadIdx.x;
  const int lane = tid & 31;
  const int w = tid >> 5;
  const int ml = lane & 15;
  const int hi = lane >> 4;
  const int klo = hi ? 8 : 0;
  const int bh = blockIdx.x;
  const int n0 = blockIdx.y * NT;

  v8f vzero = 0.0f;

  for (int d = 0; d < DD; ++d) PT[tid][d] = (__bf16)Pg[d * RR + tid];

  const float* kvb = KVws + (size_t)bh * RR * D2;
  for (int i = tid; i < RR * D2; i += 256) {
    int r = i / D2, d2 = i % D2;
    float x = kvb[i];
    if (d2 < 65) KVT[d2][r] = (__bf16)x;
  }
  const float* Qb = Qg + ((size_t)bh * NN + n0) * DD;
  for (int i = tid; i < NT * DD; i += 256) {
    int n = i / DD, d = i % DD;
    XT[n][d] = (__bf16)Qb[(size_t)n * DD + d];
  }
  __syncthreads();

  // feature matmul oriented M=r, N=n: A = P^T strip, B = Q^T (B^T layout = XT)
#pragma unroll
  for (int i = 0; i < 2; ++i) {
    int mt = 2 * w + i;
    const __bf16* arow = &PT[mt * 16 + ml][0];
    v16bf a0 = make_a_frag(arow + 0, klo);
    v16bf a1 = make_a_frag(arow + 32, klo);
    for (int nt = 0; nt < 8; ++nt) {
      const __bf16* brow = &XT[nt * 16 + ml][0];
      v16bf b0 = *(const v16bf*)(brow + hi * 16);
      v16bf b1 = *(const v16bf*)(brow + 32 + hi * 16);
      v8f c = vzero;
      c = wmma_bf16(a0, b0, c);
      c = wmma_bf16(a1, b1, c);
      v8bf f = feat8(c);
      // lane holds n = nt*16+ml fixed, r = mt*16 + 8*hi + v contiguous
      *(v8bf*)&QP[nt * 16 + ml][mt * 16 + 8 * hi] = f;
    }
  }
  __syncthreads();

  // out = Qp @ KV : wave w owns n-strip w*16..w*16+15; 5 d-tiles; K = 256
  v8f acc[5];
#pragma unroll
  for (int dt = 0; dt < 5; ++dt) acc[dt] = vzero;
#pragma unroll
  for (int kb = 0; kb < 8; ++kb) {
    v16bf a = make_a_frag(&QP[w * 16 + ml][kb * 32], klo);
#pragma unroll
    for (int dt = 0; dt < 5; ++dt) {
      v16bf b = *(const v16bf*)&KVT[dt * 16 + ml][kb * 32 + hi * 16];
      acc[dt] = wmma_bf16(a, b, acc[dt]);
    }
  }

  // norm lives in d-tile 4, column 0 (ml==0): broadcast via LDS
  if (ml == 0) {
#pragma unroll
    for (int v = 0; v < 8; ++v) NRM[w * 16 + 8 * hi + v] = acc[4][v];
  }
  __syncthreads();

  float* ob = Og + ((size_t)bh * NN + n0) * DD;
#pragma unroll
  for (int v = 0; v < 8; ++v) {
    int n = w * 16 + 8 * hi + v;
    float inv = 1.0f / fmaxf(NRM[n], 1e-6f);
#pragma unroll
    for (int dt = 0; dt < 4; ++dt)
      ob[(size_t)n * DD + dt * 16 + ml] = acc[dt][v] * inv;
  }
}

extern "C" void kernel_launch(void* const* d_in, const int* in_sizes, int n_in,
                              void* d_out, int out_size, void* d_ws, size_t ws_size,
                              hipStream_t stream) {
  const float* Q = (const float*)d_in[0];
  const float* K = (const float*)d_in[1];
  const float* V = (const float*)d_in[2];
  const float* P = (const float*)d_in[3];
  float* out = (float*)d_out;
  float* KVws = (float*)d_ws;  // BH * RR * D2 floats = 5.24 MB

  const int kvn = BH * RR * D2;
  perf_zero_kernel<<<(kvn + 255) / 256, 256, 0, stream>>>(KVws, kvn);
  perf_kv_kernel<<<dim3(BH, 8), 256, 0, stream>>>(K, V, P, KVws);
  perf_out_kernel<<<dim3(BH, NN / NT), 256, 0, stream>>>(Q, P, KVws, out);
}
